// ScaledDotProductAttention_16106127360591
// MI455X (gfx1250) — matrix operand order repaired and checked by reading, not compiled, not run on hardware
//
#include <hip/hip_runtime.h>
#include <hip/hip_bf16.h>

typedef __attribute__((ext_vector_type(16))) _Float16 v16h;
typedef __attribute__((ext_vector_type(8)))  _Float16 v8h;
typedef __attribute__((ext_vector_type(4)))  _Float16 v4h;
typedef __attribute__((ext_vector_type(8)))  float    v8f;
typedef __attribute__((ext_vector_type(4)))  float    f4;

static constexpr int S_LEN = 2048;
static constexpr int DK    = 32;
static constexpr int HNUM  = 12;
static constexpr int BNUM  = 4;
static constexpr int WAVES = 8;      // 8 waves * 16 q-rows = 128 q-rows per block
static constexpr int KPAD  = 40;     // sK row stride (halves): 80B, 16B-aligned, conflict-free
static constexpr int VPAD  = 2056;   // sVt row stride (halves): 4112B, 16B-aligned

// log2-domain: scores = (Q.K/sqrt(32))*log2(e); softmax via v_exp_f32 (=exp2)
static constexpr float QSC   = 0.17677669529663687f * 1.4426950408889634f;
static constexpr float MASKB = -1.4426950408889634e9f;   // -1e9 * log2(e)

// two aligned 16B LDS loads -> one 16-element f16 WMMA operand
__device__ __forceinline__ v16h lds2x8(const _Float16* p, int eltOff) {
  v8h lo = *(const v8h*)(p);
  v8h hi = *(const v8h*)(p + eltOff);
  return __builtin_shufflevector(lo, hi, 0,1,2,3,4,5,6,7,8,9,10,11,12,13,14,15);
}

__global__ __launch_bounds__(WAVES * 32)
void fa_fwd_kernel(const float* __restrict__ Q,
                   const float* __restrict__ K,
                   const float* __restrict__ V,
                   const unsigned char* __restrict__ mask,  // [B,1,S,S] bool (1 byte)
                   float* __restrict__ Out) {
  __shared__ _Float16 sK[S_LEN * KPAD];              // 160 KB: K f16 row-major, padded
  __shared__ _Float16 sVt[DK * VPAD];                // 128.5 KB: V^T f16, padded
  __shared__ alignas(32) float sBr[WAVES * 16];      // per-wave row-broadcast scratch

  const int lane  = threadIdx.x & 31;
  const int wave  = threadIdx.x >> 5;
  const int lhalf = lane >> 4;                       // 0 or 1
  const int lmod  = lane & 15;

  const int bh = blockIdx.y;                         // 0..47
  const int b  = bh / HNUM;
  const int q0 = (blockIdx.x * WAVES + wave) * 16;

  const size_t base = (size_t)bh * S_LEN * DK;

  // ================= stage K and V^T into LDS (f16), once per block ============
  {
    const float* Kg = K + base;
    for (int idx = threadIdx.x * 4; idx < S_LEN * DK; idx += WAVES * 32 * 4) {
      int row = idx >> 5, col = idx & 31;
      f4 f = *(const f4*)(Kg + idx);
      v4h h; h[0] = (_Float16)f.x; h[1] = (_Float16)f.y;
             h[2] = (_Float16)f.z; h[3] = (_Float16)f.w;
      *(v4h*)&sK[row * KPAD + col] = h;
    }
    const float* Vg = V + base;
    for (int idx = threadIdx.x * 4; idx < S_LEN * DK; idx += WAVES * 32 * 4) {
      int key = idx >> 5, d = idx & 31;
      f4 f = *(const f4*)(Vg + idx);
      sVt[(d + 0) * VPAD + key] = (_Float16)f.x;
      sVt[(d + 1) * VPAD + key] = (_Float16)f.y;
      sVt[(d + 2) * VPAD + key] = (_Float16)f.z;
      sVt[(d + 3) * VPAD + key] = (_Float16)f.w;
    }
  }
  __syncthreads();

  // ====== Q^T as B-matrix (32x16), pre-scaled by QSC: lane q=lmod, d=lhalf*16+e
  v16h qb;
  {
    const float* qp = Q + base + (size_t)(q0 + lmod) * DK + lhalf * 16;
    f4 a = *(const f4*)(qp);
    f4 c = *(const f4*)(qp + 4);
    f4 e = *(const f4*)(qp + 8);
    f4 g = *(const f4*)(qp + 12);
    qb[0]  = (_Float16)(a.x*QSC); qb[1]  = (_Float16)(a.y*QSC);
    qb[2]  = (_Float16)(a.z*QSC); qb[3]  = (_Float16)(a.w*QSC);
    qb[4]  = (_Float16)(c.x*QSC); qb[5]  = (_Float16)(c.y*QSC);
    qb[6]  = (_Float16)(c.z*QSC); qb[7]  = (_Float16)(c.w*QSC);
    qb[8]  = (_Float16)(e.x*QSC); qb[9]  = (_Float16)(e.y*QSC);
    qb[10] = (_Float16)(e.z*QSC); qb[11] = (_Float16)(e.w*QSC);
    qb[12] = (_Float16)(g.x*QSC); qb[13] = (_Float16)(g.y*QSC);
    qb[14] = (_Float16)(g.z*QSC); qb[15] = (_Float16)(g.w*QSC);
  }

  // all-ones B matrix (32x16) for WMMA row-sum accumulation
  v16h onesB;
#pragma unroll
  for (int e = 0; e < 16; ++e) onesB[e] = (_Float16)1.0f;

  const unsigned char* mrow = mask + ((size_t)b * S_LEN + (q0 + lmod)) * S_LEN;

  v8f o0 = {};            // O[:, 0:16)   rows q = v + 8*lhalf, col d = lmod
  v8f o1 = {};            // O[:, 16:32)
  v8f ls = {};            // row-sum accumulator, same row layout as o0/o1
  v8f cvec = {};          // C operand for score WMMAs = -m (per-lane q = lmod)

  // ============================ main loop: 64 keys / iter ======================
  for (int kc = 0; kc < S_LEN; kc += 64) {
    // ---- S^T tiles with -m pre-folded via C: s = K.Q^T*QSC - m
    v8f s[4];
#pragma unroll
    for (int i = 0; i < 4; ++i) {
      const _Float16* kp = &sK[(kc + 16 * i + lmod) * KPAD + lhalf * 16];
      v16h ka = lds2x8(kp, 8);
      s[i] = __builtin_amdgcn_wmma_f32_16x16x32_f16(false, ka, false, qb,
                                                    (short)0, cvec, false, false);
    }

    // ---- row max of (unmasked) relative scores; valid since m>=rowmax suffices
    float mloc = -INFINITY;
#pragma unroll
    for (int v = 0; v < 8; ++v)
      mloc = fmaxf(fmaxf(fmaxf(s[0][v], s[1][v]), fmaxf(s[2][v], s[3][v])), mloc);
    mloc = fmaxf(mloc, __shfl_xor(mloc, 16));

    // ---- rescale only when some row's max grew (rare after warm-up)
    if (__any(mloc > 0.0f)) {
      float du = fmaxf(mloc, 0.0f);                // per-lane (q = lmod) max increase
      sBr[wave * 16 + lmod] = du;                  // per-wave scratch; DS in-order
      v8f duv = *(const v8f*)&sBr[wave * 16 + 8 * lhalf];
#pragma unroll
      for (int v = 0; v < 8; ++v) {
        float f = __builtin_amdgcn_exp2f(-duv[v]);
        o0[v] *= f; o1[v] *= f; ls[v] *= f;
      }
#pragma unroll
      for (int v = 0; v < 8; ++v) {
        cvec[v] -= du;
        s[0][v] -= du; s[1][v] -= du; s[2][v] -= du; s[3][v] -= du;
      }
    }

    // ---- p = exp2(s + MASKB*mask_byte): one FMA + one exp per score
    v16h pa0, pa1;                                 // P in A-layout
#pragma unroll
    for (int i = 0; i < 4; ++i) {
      uint2 mb = *(const uint2*)(mrow + kc + 16 * i + lhalf * 8);
#pragma unroll
      for (int v = 0; v < 4; ++v) {
        float plo = __builtin_amdgcn_exp2f(fmaf((float)((mb.x >> (8 * v)) & 0xffu), MASKB, s[i][v]));
        float phi = __builtin_amdgcn_exp2f(fmaf((float)((mb.y >> (8 * v)) & 0xffu), MASKB, s[i][v + 4]));
        if (i < 2) { if (i == 0) { pa0[v] = (_Float16)plo; pa0[v + 4] = (_Float16)phi; }
                     else        { pa0[v + 8] = (_Float16)plo; pa0[v + 12] = (_Float16)phi; } }
        else       { if (i == 2) { pa1[v] = (_Float16)plo; pa1[v + 4] = (_Float16)phi; }
                     else        { pa1[v + 8] = (_Float16)plo; pa1[v + 12] = (_Float16)phi; } }
      }
    }

    // ---- row sums on the matrix pipe: ls += P * ones
    ls = __builtin_amdgcn_wmma_f32_16x16x32_f16(false, pa0, false, onesB, (short)0, ls, false, false);
    ls = __builtin_amdgcn_wmma_f32_16x16x32_f16(false, pa1, false, onesB, (short)0, ls, false, false);

    // ---- V B-tiles from sVt: lane = column d, table-order keys (8h+s, 16+8h+s-8)
    const _Float16* vpl = &sVt[lmod * VPAD + kc + lhalf * 8];
    const _Float16* vph = vpl + 16 * VPAD;
    v16h vb00 = lds2x8(vpl,      16);   // keys kc   ..+31, d 0..15
    v16h vb01 = lds2x8(vph,      16);   // keys kc   ..+31, d 16..31
    v16h vb10 = lds2x8(vpl + 32, 16);   // keys kc+32..+63, d 0..15
    v16h vb11 = lds2x8(vph + 32, 16);   // keys kc+32..+63, d 16..31

    // ---- O += P * V   (4x WMMA, f32 accumulate)
    o0 = __builtin_amdgcn_wmma_f32_16x16x32_f16(false, pa0, false, vb00, (short)0, o0, false, false);
    o0 = __builtin_amdgcn_wmma_f32_16x16x32_f16(false, pa1, false, vb10, (short)0, o0, false, false);
    o1 = __builtin_amdgcn_wmma_f32_16x16x32_f16(false, pa0, false, vb01, (short)0, o1, false, false);
    o1 = __builtin_amdgcn_wmma_f32_16x16x32_f16(false, pa1, false, vb11, (short)0, o1, false, false);
  }

  // ===== epilogue: row sums already live per-element in ls (same rows as O) ====
#pragma unroll
  for (int v = 0; v < 8; ++v) {
    float inv = 1.0f / ls[v];
    size_t row = (size_t)bh * S_LEN + q0 + v + 8 * lhalf;
    Out[row * DK + lmod]      = o0[v] * inv;
    Out[row * DK + lmod + 16] = o1[v] * inv;
  }
}

extern "C" void kernel_launch(void* const* d_in, const int* in_sizes, int n_in,
                              void* d_out, int out_size, void* d_ws, size_t ws_size,
                              hipStream_t stream) {
  (void)in_sizes; (void)n_in; (void)out_size; (void)d_ws; (void)ws_size;
  const float* Q = (const float*)d_in[0];
  const float* K = (const float*)d_in[1];
  const float* V = (const float*)d_in[2];
  const unsigned char* mask = (const unsigned char*)d_in[3];
  float* Out = (float*)d_out;

  dim3 grid(S_LEN / (16 * WAVES), BNUM * HNUM);    // (16, 48)
  dim3 block(WAVES * 32);                          // 256 threads = 8 waves
  fa_fwd_kernel<<<grid, block, 0, stream>>>(Q, K, V, mask, Out);
}
